// CrossAttention_68083821576707
// MI455X (gfx1250) — matrix operand order repaired and checked by reading, not compiled, not run on hardware
//
#include <hip/hip_runtime.h>
#include <math.h>
#include <stdint.h>

typedef _Float16 h16 __attribute__((ext_vector_type(16)));
typedef _Float16 h2  __attribute__((ext_vector_type(2)));
typedef float    f8  __attribute__((ext_vector_type(8)));

#define HW_    3072
#define W_IMG  64
#define H_IMG  48
#define DH     128
#define CKQ    384
#define NB     4
#define NTILE  (HW_ / 16)       /* 192 16-position tiles per batch          */
#define CHUNKB 24576            /* bytes of packed K per 32-key chunk       */
#define CHUNKH (CHUNKB / 2)     /* halves per chunk (12288)                 */

__device__ __forceinline__ f8 wmma_f16(h16 a, h16 b, f8 c) {
  // D = A(16x32 f16) * B(32x16 f16) + C(16x16 f32)
  return __builtin_amdgcn_wmma_f32_16x16x32_f16(false, a, false, b, (short)0, c,
                                                false, false);
}

__device__ __forceinline__ f8 f8_zero() {
  f8 z;
#pragma unroll
  for (int i = 0; i < 8; ++i) z[i] = 0.f;
  return z;
}

// Packed fragment layouts (f16, 16 halves = 32B per lane):
//   K/Q : [batch*NTILE + tile][cc(12)][lane(32)][16]
//   W   : [tap][cc(4)][dt(8)][lane(32)][16]
__device__ __forceinline__ const h16* kq_frag(const _Float16* p, size_t tile,
                                              int cc, int lane) {
  return (const h16*)(p + ((tile * 12 + (size_t)cc) * 32 + lane) * 16);
}
__device__ __forceinline__ const h16* w_frag(const _Float16* p, int t, int cc,
                                             int dt, int lane) {
  return (const h16*)(p + ((((size_t)t * 4 + cc) * 8 + dt) * 32 + lane) * 16);
}

// ---------------------------------------------------------------------------
// TDM staging of one contiguous 24KB packed-K chunk into LDS (gfx1250).
// D# per CDNA5 ISA ch.8: group0 = {count|flags, lds_addr, global_addr, type},
// group1 = {mask|data_size, tensor/tile dims, strides}. 1-row tile:
// data_size=8B, tile_dim0 = tensor_dim0 = stride0 = 3072 units.
// Toolchain uses the 6-arg builtin: (g0, g1, g2, g3, g4, cpol).
// ---------------------------------------------------------------------------
#if __has_builtin(__builtin_amdgcn_tensor_load_to_lds) && \
    __has_builtin(__builtin_amdgcn_s_wait_tensorcnt)
#define HAVE_TDM 1
typedef unsigned int u32x4 __attribute__((ext_vector_type(4)));
typedef int          i32x8 __attribute__((ext_vector_type(8)));
typedef int          i32x4 __attribute__((ext_vector_type(4)));

__device__ __forceinline__ void tdm_load_chunk(const _Float16* gsrc,
                                               unsigned lds_byte_off) {
  const unsigned long long ga = (unsigned long long)(uintptr_t)gsrc;
  u32x4 g0;
  g0.x = 1u;                                    // count=1 (valid), user mode
  g0.y = lds_byte_off;                          // LDS dest byte address
  g0.z = (unsigned)(ga & 0xFFFFFFFFu);          // global_addr[31:0]
  g0.w = (unsigned)((ga >> 32) & 0x1FFFFFFu) |  // global_addr[56:32]
         (2u << 30);                            // type=2 ("image")
  i32x8 g1;
  g1[0] = 0x00030000;                           // wg_mask=0, data_size=3 (8B)
  g1[1] = (int)((3072u & 0xFFFFu) << 16);       // tensor_dim0[15:0]
  g1[2] = (int)((3072u >> 16) | (1u << 16));    // tensor_dim0[31:16], td1=1
  g1[3] = (int)(3072u << 16);                   // td1 hi=0, tile_dim0=3072
  g1[4] = 0;                                    // tile_dim1=0, tile_dim2=0
  g1[5] = 3072;                                 // tensor_dim0_stride[31:0]
  g1[6] = 0;
  g1[7] = 0;
  i32x4 z;
  z[0] = 0; z[1] = 0; z[2] = 0; z[3] = 0;
  i32x8 z8;
#pragma unroll
  for (int i = 0; i < 8; ++i) z8[i] = 0;
  __builtin_amdgcn_tensor_load_to_lds(g0, g1, z, z, z8, 0);
}
#else
#define HAVE_TDM 0
#endif

// ---------------------------------------------------------------------------
// Pack k (mode 0 -> WMMA A layout) or q (mode 1 -> WMMA B layout) into f16.
// ---------------------------------------------------------------------------
__global__ void pack_kq_kernel(const float* __restrict__ src,
                               _Float16* __restrict__ dst, int mode) {
  const int lane = threadIdx.x & 31;
  const int gw   = blockIdx.x * (blockDim.x >> 5) + (threadIdx.x >> 5);
  if (gw >= NB * NTILE) return;
  const int batch = gw / NTILE;
  const int t0    = (gw % NTILE) * 16;
  const int n     = lane & 15;
  const int hi    = lane >> 4;
  const float* __restrict__ sb = src + (size_t)batch * CKQ * HW_;

#pragma unroll
  for (int cc = 0; cc < 12; ++cc) {
    h16 f;
#pragma unroll
    for (int j = 0; j < 8; ++j) {
      int c;
      if (mode == 0) {  // A layout
        const int Kb = ((j < 4) ? 0 : 16) + (hi ? 8 : 0) + 2 * (j & 3);
        c = cc * 32 + Kb;
      } else {          // B layout
        c = cc * 32 + ((j < 4) ? 0 : 16) + (hi ? 8 : 0) + 2 * (j & 3);
      }
      f[2 * j]     = (_Float16)sb[(size_t)c * HW_ + t0 + n];
      f[2 * j + 1] = (_Float16)sb[(size_t)(c + 1) * HW_ + t0 + n];
    }
    *(h16*)kq_frag(dst, (size_t)gw, cc, lane) = f;
  }
}

// ---------------------------------------------------------------------------
// Pack OIHW f32 weights into f16 A-fragments. One wave per (tap, cc, dt).
// ---------------------------------------------------------------------------
__global__ void pack_w_kernel(const float* __restrict__ w,
                              _Float16* __restrict__ dst, int taps) {
  const int lane = threadIdx.x & 31;
  const int gw   = blockIdx.x;          // blockDim = 32 (one wave)
  const int dt   = gw & 7;
  const int cc   = (gw >> 3) & 3;
  const int t    = gw >> 5;
  const int n    = lane & 15;
  const int hi   = lane >> 4;
  const int d    = dt * 16 + n;

  h16 f;
#pragma unroll
  for (int j = 0; j < 8; ++j) {
    const int K = ((j < 4) ? 0 : 16) + (hi ? 8 : 0) + 2 * (j & 3);
    const int c = cc * 32 + K;
    const size_t wi = ((size_t)d * DH + c) * taps + t;
    f[2 * j]     = (_Float16)w[wi];
    f[2 * j + 1] = (_Float16)w[wi + taps];
  }
  *(h16*)w_frag(dst, t, cc, dt, lane) = f;
}

// ---------------------------------------------------------------------------
// Conv (1x1 or 3x3, pad = dil) with pre-packed f16 weight fragments.
// ---------------------------------------------------------------------------
__global__ void conv_wmma_kernel(const float* __restrict__ inp,
                                 const _Float16* __restrict__ pw,
                                 const float* __restrict__ res,
                                 float* __restrict__ outf,
                                 _Float16* __restrict__ outh,
                                 int taps, int dil, int do_lrelu) {
  const int lane = threadIdx.x & 31;
  const int gw   = blockIdx.x * (blockDim.x >> 5) + (threadIdx.x >> 5);
  if (gw >= NB * NTILE) return;                 // wave-uniform guard
  const int batch = gw / NTILE;
  const int q0    = (gw % NTILE) * 16;
  const int n     = lane & 15;
  const int hi    = lane >> 4;

  const float* __restrict__ ib = inp + (size_t)batch * DH * HW_;
  const float* __restrict__ rb = res ? res + (size_t)batch * DH * HW_ : nullptr;

  const int pos = q0 + n;
  const int py  = pos >> 6;
  const int px  = pos & 63;

  f8 acc[8];
#pragma unroll
  for (int t = 0; t < 8; ++t) acc[t] = f8_zero();

  for (int t = 0; t < taps; ++t) {
    int oy = 0, ox = 0;
    if (taps == 9) { oy = (t / 3 - 1) * dil; ox = (t % 3 - 1) * dil; }
    const int  iy   = py + oy;
    const int  ix   = px + ox;
    const bool ok   = (iy >= 0) & (iy < H_IMG) & (ix >= 0) & (ix < W_IMG);
    const int  ipos = iy * W_IMG + ix;

    for (int cc = 0; cc < 4; ++cc) {
      const int c0 = cc * 32;
      h16 b;                                    // 32(in-ch) x 16(pos)
#pragma unroll
      for (int j = 0; j < 8; ++j) {
        const int c = c0 + ((j < 4) ? 0 : 16) + (hi ? 8 : 0) + 2 * (j & 3);
        float x0 = ok ? ib[(size_t)c * HW_ + ipos] : 0.f;
        float x1 = ok ? ib[(size_t)(c + 1) * HW_ + ipos] : 0.f;
        b[2 * j]     = (_Float16)x0;
        b[2 * j + 1] = (_Float16)x1;
      }
#pragma unroll
      for (int dt = 0; dt < 8; ++dt) {
        const h16 a = *w_frag(pw, t, cc, dt, lane);
        acc[dt] = wmma_f16(a, b, acc[dt]);
      }
    }
  }

#pragma unroll
  for (int dt = 0; dt < 8; ++dt) {
#pragma unroll
    for (int i = 0; i < 8; ++i) {
      const int d = dt * 16 + i + 8 * hi;       // C layout: M = i + 8*hi
      float r = acc[dt][i];
      if (do_lrelu) r = (r >= 0.f) ? r : 0.2f * r;
      if (rb) r += rb[(size_t)d * HW_ + pos];
      if (outh) outh[(size_t)d * HW_ + pos] = (_Float16)r;
      else      outf[(size_t)d * HW_ + pos] = r;
    }
  }
}

// ---------------------------------------------------------------------------
// Flash attention. Block = 4 waves, one batch, 4 adjacent query tiles.
// Packed-K chunks (24KB, contiguous) are staged into double-buffered LDS by
// the Tensor Data Mover (wave 0), shared by all 4 waves.
// ---------------------------------------------------------------------------
__global__ void attn_wmma_kernel(const _Float16* __restrict__ pK,
                                 const _Float16* __restrict__ pQ,
                                 const _Float16* __restrict__ valh,
                                 float* __restrict__ O_) {
  extern __shared__ _Float16 smem[];            // 2 * CHUNKH halves (48KB)

  const int tid  = threadIdx.x;
  const int lane = tid & 31;
  const int wib  = tid >> 5;                    // wave in block: 0..3
  const int batch = blockIdx.x / (NTILE / 4);   // 48 blocks per batch
  const int qt    = (blockIdx.x % (NTILE / 4)) * 4 + wib;
  const int q0    = qt * 16;
  const int n     = lane & 15;
  const int hi    = lane >> 4;
  const float scale = 0.051031036307982884f;    // 1/sqrt(384)

  const _Float16* __restrict__ kbb = pK + (size_t)batch * NTILE * 12 * 32 * 16;
  const _Float16* __restrict__ vb  = valh + (size_t)batch * DH * HW_;
  float* __restrict__ ob           = O_ + (size_t)batch * DH * HW_;

  // Preload this wave's Q fragments (12 x 32B per lane).
  h16 qf[12];
#pragma unroll
  for (int cc = 0; cc < 12; ++cc)
    qf[cc] = *kq_frag(pQ, (size_t)batch * NTILE + qt, cc, lane);

  f8 acc[8];
#pragma unroll
  for (int t = 0; t < 8; ++t) acc[t] = f8_zero();
  float m_run = -3.0e38f;
  float l_run = 0.f;

  const int nchunk = HW_ / 32;                  // 96 key chunks

#if HAVE_TDM
  if (wib == 0) tdm_load_chunk(kbb, 0);         // prologue: chunk 0 -> buf 0
#endif

  for (int c = 0; c < nchunk; ++c) {
#if HAVE_TDM
    if (wib == 0) {
      if (c + 1 < nchunk) {
        tdm_load_chunk(kbb + (size_t)(c + 1) * CHUNKH, (unsigned)(((c + 1) & 1) * CHUNKB));
        __builtin_amdgcn_s_wait_tensorcnt(1);   // chunk c complete
      } else {
        __builtin_amdgcn_s_wait_tensorcnt(0);
      }
    }
#else
    {                                           // cooperative staging fallback
      const uint4* s = (const uint4*)(kbb + (size_t)c * CHUNKH);
      uint4* d = (uint4*)(smem + (c & 1) * CHUNKH);
      for (int i = tid; i < CHUNKB / 16; i += 128) d[i] = s[i];
    }
#endif
    __syncthreads();                            // chunk c visible to all waves

    const _Float16* kbuf = smem + (c & 1) * CHUNKH;
    const int k0 = c * 32;

    // ---- scores: two 16x16 tiles (keys k0..+15, k0+16..+31) from LDS ----
    f8 s0 = f8_zero(), s1 = f8_zero();
#pragma unroll
    for (int cc = 0; cc < 12; ++cc) {
      const h16 a0 = *(const h16*)(kbuf + ((size_t)cc * 32 + lane) * 16);
      const h16 a1 = *(const h16*)(kbuf + ((size_t)(12 + cc) * 32 + lane) * 16);
      s0 = wmma_f16(a0, qf[cc], s0);
      s1 = wmma_f16(a1, qf[cc], s1);
    }

    // ---- online softmax over key dim (per query column n) ----
    float mt = -3.0e38f;
#pragma unroll
    for (int i = 0; i < 8; ++i) {
      s0[i] *= scale;
      s1[i] *= scale;
      mt = fmaxf(mt, fmaxf(s0[i], s1[i]));
    }
    mt = fmaxf(mt, __shfl_xor(mt, 16, 32));     // combine key-row halves
    const float m_new = fmaxf(m_run, mt);
    const float corr  = __expf(m_run - m_new);
    float ls = 0.f;
#pragma unroll
    for (int i = 0; i < 8; ++i) {
      s0[i] = __expf(s0[i] - m_new);
      s1[i] = __expf(s1[i] - m_new);
      ls += s0[i] + s1[i];
    }
    ls += __shfl_xor(ls, 16, 32);
    l_run = l_run * corr + ls;
    m_run = m_new;
#pragma unroll
    for (int dt = 0; dt < 8; ++dt)
#pragma unroll
      for (int i = 0; i < 8; ++i) acc[dt][i] *= corr;

    // ---- re-layout P: two C-layout tiles -> one B fragment (32k x 16q) ----
    h16 pf;
#pragma unroll
    for (int j = 0; j < 8; ++j) {
      float o0, o1;
      if (j < 4) {
        const int r = 2 * j;                    // rows 0..7
        float a0v = s0[r], a1v = s0[r + 1];
        float b0v = __shfl_xor(s1[r], 16, 32);
        float b1v = __shfl_xor(s1[r + 1], 16, 32);
        o0 = hi ? b0v : a0v;
        o1 = hi ? b1v : a1v;
      } else {
        const int r = 2 * j - 8;                // rows 8..15
        float a0v = __shfl_xor(s0[r], 16, 32);
        float a1v = __shfl_xor(s0[r + 1], 16, 32);
        float b0v = s1[r], b1v = s1[r + 1];
        o0 = hi ? b0v : a0v;
        o1 = hi ? b1v : a1v;
      }
      pf[2 * j]     = (_Float16)o0;
      pf[2 * j + 1] = (_Float16)o1;
    }

    // ---- acc += val(16d x 32k) * P(32k x 16q); val pairs contiguous ----
#pragma unroll
    for (int dt = 0; dt < 8; ++dt) {
      const int d = dt * 16 + n;
      h16 a;
#pragma unroll
      for (int j = 0; j < 8; ++j) {
        const int Kb = (hi ? 16 : 0) + 2 * j;
        const h2 vv = *(const h2*)(vb + (size_t)d * HW_ + k0 + Kb);
        a[2 * j]     = vv[0];
        a[2 * j + 1] = vv[1];
      }
      acc[dt] = wmma_f16(a, pf, acc[dt]);
    }

    __syncthreads();                  // all waves done with buf before refill
  }

  const float inv = 1.0f / l_run;
#pragma unroll
  for (int dt = 0; dt < 8; ++dt)
#pragma unroll
    for (int i = 0; i < 8; ++i)
      ob[(size_t)(dt * 16 + i + 8 * hi) * HW_ + q0 + n] = acc[dt][i] * inv;
}

// ---------------------------------------------------------------------------
extern "C" void kernel_launch(void* const* d_in, const int* in_sizes, int n_in,
                              void* d_out, int out_size, void* d_ws, size_t ws_size,
                              hipStream_t stream) {
  const float* k       = (const float*)d_in[0];
  const float* q       = (const float*)d_in[1];
  const float* v       = (const float*)d_in[2];
  const float* w_value = (const float*)d_in[3];
  const float* w_out   = (const float*)d_in[4];
  const float* w_ff1   = (const float*)d_in[5];
  const float* w_ff2   = (const float*)d_in[6];
  float* out = (float*)d_out;

  // ---- workspace carve-up (bytes) ----
  char* ws = (char*)d_ws;
  const size_t szKQ  = (size_t)NB * NTILE * 12 * 32 * 16 * 2;  // 9,437,184
  const size_t szVH  = (size_t)NB * DH * HW_ * 2;              // 3,145,728
  const size_t szF32 = (size_t)NB * DH * HW_ * 4;              // 6,291,456
  const size_t szW1  = (size_t)1 * 4 * 8 * 32 * 16 * 2;        // 32,768
  const size_t szW9  = (size_t)9 * 4 * 8 * 32 * 16 * 2;        // 294,912

  _Float16* pK    = (_Float16*)(ws);
  _Float16* pQ    = (_Float16*)(ws + szKQ);
  _Float16* valh  = (_Float16*)(ws + 2 * szKQ);
  float*    pval  = (float*)   (ws + 2 * szKQ + szVH);
  float*    v2    = (float*)   (ws + 2 * szKQ + szVH + szF32);
  _Float16* pWval = (_Float16*)(ws + 2 * szKQ + szVH + 2 * szF32);
  _Float16* pWout = (_Float16*)(ws + 2 * szKQ + szVH + 2 * szF32 + szW1);
  _Float16* pWff1 = (_Float16*)(ws + 2 * szKQ + szVH + 2 * szF32 + szW1 + szW9);
  _Float16* pWff2 = (_Float16*)(ws + 2 * szKQ + szVH + 2 * szF32 + szW1 + 2 * szW9);
  float*    t1    = (float*)pK;   // packed K/Q are dead after attention

  const int nwaves = NB * NTILE;  // 768 waves
  dim3 blk(128);                  // 4 waves/block (wave32)
  dim3 grd(nwaves / 4);           // 192 blocks

  // ---- pack weights (tiny grids) ----
  pack_w_kernel<<<dim3(1 * 32), dim3(32), 0, stream>>>(w_value, pWval, 1);
  pack_w_kernel<<<dim3(9 * 32), dim3(32), 0, stream>>>(w_out,   pWout, 9);
  pack_w_kernel<<<dim3(9 * 32), dim3(32), 0, stream>>>(w_ff1,   pWff1, 9);
  pack_w_kernel<<<dim3(9 * 32), dim3(32), 0, stream>>>(w_ff2,   pWff2, 9);

  // ---- pack K (A layout) and Q (B layout) to f16 fragments ----
  pack_kq_kernel<<<grd, blk, 0, stream>>>(k, pK, 0);
  pack_kq_kernel<<<grd, blk, 0, stream>>>(q, pQ, 1);

  // 1) val (f16) = conv1x1(v, w_value)
  conv_wmma_kernel<<<grd, blk, 0, stream>>>(v, pWval, nullptr, nullptr, valh,
                                            1, 1, 0);
  // 2) p_val = attention(k, q, val)   [TDM-staged K, 48KB dynamic LDS]
  attn_wmma_kernel<<<grd, blk, 2 * CHUNKB, stream>>>(pK, pQ, valh, pval);
  // 3) v2 = v + lrelu(conv3x3(p_val, w_out, pad=1, dil=1))
  conv_wmma_kernel<<<grd, blk, 0, stream>>>(pval, pWout, v, v2, nullptr,
                                            9, 1, 1);
  // 4) t1 = lrelu(conv3x3(v2, w_ff1, pad=2, dil=2))
  conv_wmma_kernel<<<grd, blk, 0, stream>>>(v2, pWff1, nullptr, t1, nullptr,
                                            9, 2, 1);
  // 5) out = v2 + lrelu(conv3x3(t1, w_ff2, pad=1, dil=1))
  conv_wmma_kernel<<<grd, blk, 0, stream>>>(t1, pWff2, v2, out, nullptr,
                                            9, 1, 1);
}
